// LSTMEmbed_82867099009919
// MI455X (gfx1250) — compile-verified, matrix operand order checked by reading, not run on hardware
//
#include <hip/hip_runtime.h>
#include <hip/hip_bf16.h>
#include <cstdint>

// ---------------------------------------------------------------------------
// LSTMEmbed on MI455X (gfx1250): embedding + input GEMM (bf16 WMMA),
// LDS-resident recurrent scan (bf16 WMMA, W_hh pinned in LDS), final gather.
// ---------------------------------------------------------------------------

#define Bn 64
#define Tt 2048
#define Dh 128
#define Gg 512     // 4*Dh
#define Vv 5000
#define Nn 512

typedef __attribute__((ext_vector_type(16))) __bf16 v16bf;
typedef __attribute__((ext_vector_type(8)))  float  v8f;

union Frag {
    v16bf v;
    float4 f4[2];
};

// -------------------------- workspace layout (bytes) -----------------------
#define OFF_W2V   ((size_t)0)                       // V*Dh bf16   = 1,280,000
#define OFF_WIH   ((size_t)1280000)                 // G*Dh bf16   =   131,072
#define OFF_WHH   ((size_t)1411072)                 // G*Dh bf16   =   131,072
#define OFF_BSUM  ((size_t)1542144)                 // G f32       =     2,048
#define OFF_XG    ((size_t)1544192)                 // B*T*G f32   = 268,435,456
#define OFF_HS    ((size_t)269979648)               // B*T*Dh f32  =  67,108,864
#define WS_NEED   ((size_t)337088512)

// -------------------- kernel 0: weight conversion to bf16 ------------------
__global__ __launch_bounds__(256) void k_convert(
    const float* __restrict__ w2v, const float* __restrict__ wih,
    const float* __restrict__ whh, const float* __restrict__ b_ih,
    const float* __restrict__ b_hh,
    __bf16* __restrict__ w2v_bf, __bf16* __restrict__ wih_bf,
    __bf16* __restrict__ whh_bf, float* __restrict__ bsum) {
    int i = blockIdx.x * 256 + threadIdx.x;
    if (i < Vv * Dh) w2v_bf[i] = (__bf16)w2v[i];
    if (i < Gg * Dh) {
        wih_bf[i] = (__bf16)wih[i];
        whh_bf[i] = (__bf16)whh[i];
    }
    if (i < Gg) bsum[i] = b_ih[i] + b_hh[i];
}

// ---- kernel 1: x_gates = gather(w2v, token_idx) @ W_ih^T + (b_ih+b_hh) ----
// grid.x = (B*T)/16 M-tiles; 8 waves/block, each wave owns 4 N-tiles of 16.
__global__ __launch_bounds__(256) void k_xgates(
    const int* __restrict__ token_idx, const __bf16* __restrict__ w2v_bf,
    const __bf16* __restrict__ wih_bf, const float* __restrict__ bsum,
    float* __restrict__ xg) {
    const int mbase = blockIdx.x * 16;          // row tile of [B*T]
    const int lane  = threadIdx.x & 31;
    const int wave  = threadIdx.x >> 5;
    const int lr    = lane & 15;
    const int hi    = lane >> 4;

    // A fragments: one vocab row per M index; both lane halves cover M=lr.
    const int vrow = token_idx[mbase + lr];
    const __bf16* ap = w2v_bf + (size_t)vrow * Dh;
    Frag a[4];
#pragma unroll
    for (int c = 0; c < 4; ++c) {
        const int k0 = c * 32 + hi * 8;
        a[c].f4[0] = *(const float4*)(ap + k0);
        a[c].f4[1] = *(const float4*)(ap + k0 + 16);
    }

#pragma unroll
    for (int i = 0; i < 4; ++i) {
        const int nt = wave * 4 + i;
        const int g  = nt * 16 + lr;            // output gate column (N)
        const __bf16* bp = wih_bf + (size_t)g * Dh;

        v8f acc;
        const float bias = bsum[g];
#pragma unroll
        for (int r = 0; r < 8; ++r) acc[r] = bias;

#pragma unroll
        for (int c = 0; c < 4; ++c) {
            Frag bfr;
            const int k0 = c * 32 + hi * 8;
            bfr.f4[0] = *(const float4*)(bp + k0);
            bfr.f4[1] = *(const float4*)(bp + k0 + 16);
            acc = __builtin_amdgcn_wmma_f32_16x16x32_bf16(
                false, a[c].v, false, bfr.v, (short)0, acc, false, false);
        }
        // D layout: M = r + 8*hi, N = g
#pragma unroll
        for (int r = 0; r < 8; ++r)
            xg[(size_t)(mbase + hi * 8 + r) * Gg + g] = acc[r];
    }
}

// ------------- kernel 2: sequential LSTM scan, LDS-resident ----------------
// grid.x = B/16 = 4 blocks; each block owns 16 batch rows for all 2048 steps.
// Dynamic LDS: W_hh bf16 (128KB) | h bf16 (4KB) | c f32 (8KB) | gates f32 (32KB)
#define LDS_WHH   ((size_t)0)
#define LDS_H     ((size_t)131072)
#define LDS_C     ((size_t)135168)
#define LDS_GATES ((size_t)143360)
#define LDS_BYTES ((size_t)176128)

__global__ __launch_bounds__(256) void k_scan(
    const float* __restrict__ xg, const __bf16* __restrict__ whh_bf,
    float* __restrict__ hs) {
    const int b0 = blockIdx.x * 16;
    extern __shared__ char smem[];
    __bf16* whh_s   = (__bf16*)(smem + LDS_WHH);
    __bf16* h_s     = (__bf16*)(smem + LDS_H);
    float*  c_s     = (float*)(smem + LDS_C);
    float*  gates_s = (float*)(smem + LDS_GATES);

    // Pin W_hh (bf16, 128KB) in LDS for the whole scan.
    {
        const float4* src = (const float4*)whh_bf;
        float4* dst = (float4*)whh_s;
        for (int i = threadIdx.x; i < (Gg * Dh) / 8; i += 256) dst[i] = src[i];
    }
    for (int i = threadIdx.x; i < 16 * Dh; i += 256) {
        h_s[i] = (__bf16)0.0f;
        c_s[i] = 0.0f;
    }
    __syncthreads();

    const int lane = threadIdx.x & 31;
    const int wave = threadIdx.x >> 5;
    const int lr   = lane & 15;
    const int hi   = lane >> 4;

    for (int t = 0; t < Tt; ++t) {
        // ---- gates = x_gates[:,t,:] + h @ W_hh^T (bf16 WMMA, K=128) ----
        Frag a[4];
        const __bf16* ap = h_s + lr * Dh;       // A row = batch lane lr
#pragma unroll
        for (int c = 0; c < 4; ++c) {
            const int k0 = c * 32 + hi * 8;
            a[c].f4[0] = *(const float4*)(ap + k0);
            a[c].f4[1] = *(const float4*)(ap + k0 + 16);
        }
#pragma unroll
        for (int i = 0; i < 4; ++i) {
            const int nt = wave * 4 + i;
            const int g  = nt * 16 + lr;
            const __bf16* bp = whh_s + (size_t)g * Dh;

            // C init from precomputed x_gates (already includes biases).
            v8f acc;
            const float* xp =
                xg + ((size_t)(b0 + hi * 8) * Tt + t) * Gg + g;
#pragma unroll
            for (int r = 0; r < 8; ++r) acc[r] = xp[(size_t)r * Tt * Gg];
            if (t + 1 < Tt)  // hint next step's slice toward GL2
                __builtin_prefetch(xp + Gg, 0, 1);

#pragma unroll
            for (int c = 0; c < 4; ++c) {
                Frag bfr;
                const int k0 = c * 32 + hi * 8;
                bfr.f4[0] = *(const float4*)(bp + k0);
                bfr.f4[1] = *(const float4*)(bp + k0 + 16);
                acc = __builtin_amdgcn_wmma_f32_16x16x32_bf16(
                    false, a[c].v, false, bfr.v, (short)0, acc, false, false);
            }
#pragma unroll
            for (int r = 0; r < 8; ++r)
                gates_s[(hi * 8 + r) * Gg + g] = acc[r];
        }
        __syncthreads();

        // ---- elementwise gate math: i,f,g,o -> c,h ----
        for (int e = threadIdx.x; e < 16 * Dh; e += 256) {
            const int bl = e >> 7;
            const int d  = e & 127;
            float iv = gates_s[bl * Gg + d];
            float fv = gates_s[bl * Gg + Dh + d];
            float gv = gates_s[bl * Gg + 2 * Dh + d];
            float ov = gates_s[bl * Gg + 3 * Dh + d];
            iv = 1.0f / (1.0f + __expf(-iv));
            fv = 1.0f / (1.0f + __expf(-fv));
            gv = tanhf(gv);
            ov = 1.0f / (1.0f + __expf(-ov));
            const float cv = fv * c_s[e] + iv * gv;
            const float hv = ov * tanhf(cv);
            c_s[e] = cv;
            h_s[e] = (__bf16)hv;
            hs[((size_t)(b0 + bl) * Tt + t) * Dh + d] = hv;
        }
        __syncthreads();
    }
}

// ----------------- kernel 3: node gather + append final h ------------------
__global__ __launch_bounds__(256) void k_gather(
    const int* __restrict__ node_pos, const float* __restrict__ hs,
    float* __restrict__ out) {
    const size_t idx = (size_t)blockIdx.x * 256 + threadIdx.x;
    const size_t total = (size_t)Bn * (Nn + 1) * Dh;
    if (idx >= total) return;
    const int d = (int)(idx & (Dh - 1));
    const size_t r = idx >> 7;
    const int b = (int)(r / (Nn + 1));
    const int j = (int)(r % (Nn + 1));
    const int t = (j == Nn) ? (Tt - 1) : node_pos[b * Nn + j];
    out[idx] = hs[((size_t)b * Tt + t) * Dh + d];
}

// ---------------------------------------------------------------------------
extern "C" void kernel_launch(void* const* d_in, const int* in_sizes, int n_in,
                              void* d_out, int out_size, void* d_ws,
                              size_t ws_size, hipStream_t stream) {
    if (ws_size < WS_NEED) return;  // workspace too small: refuse (deterministic)

    const int*   token_idx = (const int*)d_in[0];
    const int*   node_pos  = (const int*)d_in[1];
    const float* w2v       = (const float*)d_in[2];
    const float* W_ih      = (const float*)d_in[3];
    const float* W_hh      = (const float*)d_in[4];
    const float* b_ih      = (const float*)d_in[5];
    const float* b_hh      = (const float*)d_in[6];
    float* out = (float*)d_out;

    char* ws = (char*)d_ws;
    __bf16* w2v_bf = (__bf16*)(ws + OFF_W2V);
    __bf16* wih_bf = (__bf16*)(ws + OFF_WIH);
    __bf16* whh_bf = (__bf16*)(ws + OFF_WHH);
    float*  bsum   = (float*)(ws + OFF_BSUM);
    float*  xg     = (float*)(ws + OFF_XG);
    float*  hs     = (float*)(ws + OFF_HS);

    // 0) convert weights to bf16, fold biases
    k_convert<<<(Vv * Dh + 255) / 256, 256, 0, stream>>>(
        w2v, W_ih, W_hh, b_ih, b_hh, w2v_bf, wih_bf, whh_bf, bsum);

    // 1) fused embedding gather + input GEMM -> x_gates [B*T, 4D]
    k_xgates<<<(Bn * Tt) / 16, 256, 0, stream>>>(token_idx, w2v_bf, wih_bf,
                                                 bsum, xg);

    // 2) recurrent scan, W_hh + state resident in LDS
    k_scan<<<Bn / 16, 256, (uint32_t)LDS_BYTES, stream>>>(xg, whh_bf, hs);

    // 3) node gather + final hidden state
    const size_t total = (size_t)Bn * (Nn + 1) * Dh;
    k_gather<<<(unsigned)((total + 255) / 256), 256, 0, stream>>>(node_pos, hs,
                                                                  out);
}
